// DySnakeConv_10118942949520
// MI455X (gfx1250) — matrix operand order repaired and checked by reading, not compiled, not run on hardware
//
#include <hip/hip_runtime.h>
#include <math.h>

typedef __attribute__((ext_vector_type(16))) _Float16 v16h;
typedef __attribute__((ext_vector_type(8)))  _Float16 v8h;
typedef __attribute__((ext_vector_type(8)))  float    v8f;
typedef __attribute__((ext_vector_type(2)))  float    v2f;

#define BATCH 8
#define CH    32
#define H     384
#define W     384
#define TY    4                   // output tile rows
#define TX    32                  // output tile cols
#define EXH   (TY + 2)            // 6   extended rows (depthwise halo)
#define EXW   (TX + 2)            // 34  extended cols
#define NP    (EXH * EXW)         // 204 extended positions
#define XROWS (TY + 6)            // 10  x-tile rows (halo 3 top / 3 bottom)
#define XCOLS (TX + 6)            // 38  x-tile cols
#define XPITCH 40
#define XPLANE2 (XROWS * XPITCH * 2) // 800 floats per channel-PAIR plane
#define NTILES ((NP + 15) / 16)   // 13 WMMA M-tiles
#define NWAVES 8

#if __has_builtin(__builtin_amdgcn_global_load_async_to_lds_b32) && \
    __has_builtin(__builtin_amdgcn_s_wait_asynccnt)
#define USE_ASYNC_LDS 1
typedef __attribute__((address_space(1))) void  g_void;
typedef __attribute__((address_space(1))) int   g_int;
typedef __attribute__((address_space(3))) void  l_void;
typedef __attribute__((address_space(3))) int   l_int;
#else
#define USE_ASYNC_LDS 0
#endif

// torch grid_sample reflection (align_corners=False) + clip, as in reference
__device__ __forceinline__ float reflect_clip(float c, float size) {
  float xr    = fabsf(c + 0.5f);
  float extra = fmodf(xr, size);
  float flips = floorf(xr / size);
  float par   = fmodf(flips, 2.0f);
  float out   = (par == 0.0f) ? (extra - 0.5f) : (size - extra - 0.5f);
  return fminf(fmaxf(out, 0.0f), size - 1.0f);
}

__global__ __launch_bounds__(256) void dysnake_fused(
    const float* __restrict__ x, const float* __restrict__ offset_w,
    const float* __restrict__ offset_b, const float* __restrict__ dw_w,
    float* __restrict__ out)
{
  // x tile stored channel-pair interleaved: xs[c>>1][row][col][c&1]
  __shared__ float    xs[(CH / 2) * XPLANE2];   // 51200 B, zero-padded halo
  __shared__ float    offs[2 * NP];             //  1632 B  tanh'd offsets
  __shared__ float    xdef[NWAVES * NP];        //  6528 B  per-wave deformed tile
  __shared__ __align__(16) _Float16 wstage[9 * 2 * 2 * 16]; // 1152 B f16 weights

  const int tid  = threadIdx.x;
  const int lane = tid & 31;
  const int wave = tid >> 5;
  const int j0 = blockIdx.x * TX;
  const int i0 = blockIdx.y * TY;
  const int bb = blockIdx.z;

  // ---------------- Phase A: stage x tile (zero-padded halo) + weights ----
  for (int idx = tid; idx < CH * XROWS * XCOLS; idx += 256) {
    int c = idx / (XROWS * XCOLS);
    int r = (idx / XCOLS) % XROWS;
    int q = idx % XCOLS;
    int gy = i0 - 3 + r, gx = j0 - 3 + q;
    int didx = (c >> 1) * XPLANE2 + (r * XPITCH + q) * 2 + (c & 1);
    bool inb = (unsigned)gy < (unsigned)H && (unsigned)gx < (unsigned)W;
#if USE_ASYNC_LDS
    if (inb) {
      // CDNA5 async DMA: global -> LDS, no VGPR round trip (ASYNCcnt).
      g_int* gp = (g_int*)(g_void*)(x + ((bb * CH + c) * H + gy) * W + gx);
      l_int* lp = (l_int*)(l_void*)(xs + didx);
      __builtin_amdgcn_global_load_async_to_lds_b32(gp, lp, 0, 0);
    } else {
      xs[didx] = 0.0f;
    }
#else
    float v = 0.0f;
    if (inb) v = x[((bb * CH + c) * H + gy) * W + gx];
    xs[didx] = v;
#endif
  }
  // wstage[((tap*2 + kgroup)*2 + n)*16 + e] = offset_w[n][kgroup*16+e][tap]
  for (int idx = tid; idx < 9 * 2 * 2 * 16; idx += 256) {
    int e = idx & 15, n = (idx >> 4) & 1, kg = (idx >> 5) & 1, tap = idx >> 6;
    wstage[idx] = (_Float16)offset_w[(n * CH + kg * 16 + e) * 9 + tap];
  }
#if USE_ASYNC_LDS
  __builtin_amdgcn_s_wait_asynccnt(0);
#endif
  __syncthreads();

  // ---------------- Phase B: offset conv via v_wmma_f32_16x16x32_f16 ------
  // M = 16 extended spatial positions, K = 32 channels per tap (9 taps over
  // two independent accumulators -> no WMMA->WMMA RAW hazard NOPs),
  // N=0 -> offset-x, N=1 -> offset-y, N>=2 padded zero.
  {
    const int hi = lane >> 4;        // K-group of this lane
    const int n  = lane & 15;        // M row (for A) / N column (for B,D)
    const bool nvalid = (n < 2);

    // Build all 9 B fragments once: unconditional wide LDS loads from a
    // per-lane VALID address, then select-to-zero (no predicated loads).
    v16h bfrag[9];
    {
      const int wsel = n & 1;
      #pragma unroll
      for (int tap = 0; tap < 9; ++tap) {
        const _Float16* wp = wstage + ((tap * 2) * 2 + wsel) * 16 + hi * 8;
        v8h w0 = *(const v8h*)wp;
        v8h w1 = *(const v8h*)(wp + 32);
        #pragma unroll
        for (int e = 0; e < 8; ++e) {
          bfrag[tap][e]     = nvalid ? w0[e] : (_Float16)0.0f;
          bfrag[tap][e + 8] = nvalid ? w1[e] : (_Float16)0.0f;
        }
      }
    }

    for (int tt = wave; tt < NTILES; tt += NWAVES) {
      int pos = tt * 16 + n;
      int pc  = pos < NP ? pos : NP - 1;
      int py = pc / EXW, px = pc - py * EXW;
      v8f acc0 = {}, acc1 = {};
      #pragma unroll
      for (int tap = 0; tap < 9; ++tap) {
        int ky = tap / 3, kx = tap - ky * 3;
        // conv input at (gy+ky-1, gx+kx-1); local origin is (i0-3, j0-3)
        int cell = (py + ky + 1) * XPITCH + (px + kx + 1);
        v16h a;
        #pragma unroll
        for (int ep = 0; ep < 8; ++ep) {
          // channel pair index for A elements {2ep, 2ep+1}
          int pr = ep + (ep & 4) + (hi << 2);
          v2f v2 = *(const v2f*)(xs + pr * XPLANE2 + cell * 2);
          a[2 * ep]     = (_Float16)v2.x;
          a[2 * ep + 1] = (_Float16)v2.y;
        }
        if (tap & 1)
          acc1 = __builtin_amdgcn_wmma_f32_16x16x32_f16(
              false, a, false, bfrag[tap], (short)0, acc1, false, false);
        else
          acc0 = __builtin_amdgcn_wmma_f32_16x16x32_f16(
              false, a, false, bfrag[tap], (short)0, acc0, false, false);
      }
      if (nvalid) {                  // D: VGPR j = row M=j (+8 for hi lanes)
        float bias = offset_b[n];
        int mbase = hi << 3;
        #pragma unroll
        for (int j = 0; j < 8; ++j) {
          int p2 = tt * 16 + mbase + j;
          if (p2 < NP)
            offs[n * NP + p2] = tanhf(acc0[j] + acc1[j] + bias);
        }
      }
    }
  }
  __syncthreads();

  // ---------------- Phase C: grid-sample + depthwise conv + ReLU ----------
  for (int cc = 0; cc < 4; ++cc) {
    int c = cc * NWAVES + wave;          // one channel per wave
    const float* xp = xs + (c >> 1) * XPLANE2 + (c & 1);
    float w9[9];
    #pragma unroll
    for (int t = 0; t < 9; ++t) w9[t] = dw_w[c * 9 + t];

    for (int t = lane; t < NP; t += 32) {
      int py = t / EXW, px = t - py * EXW;
      int gy = i0 - 1 + py, gx = j0 - 1 + px;
      float v = 0.0f;                    // depthwise conv zero padding
      if ((unsigned)gy < (unsigned)H && (unsigned)gx < (unsigned)W) {
        float fx = (float)gx + offs[t];
        float fy = (float)gy + offs[NP + t];
        float gxn = 2.0f * fx / (float)(W - 1) - 1.0f;
        float gyn = 2.0f * fy / (float)(H - 1) - 1.0f;
        float ix = ((gxn + 1.0f) * (float)W - 1.0f) * 0.5f;
        float iy = ((gyn + 1.0f) * (float)H - 1.0f) * 0.5f;
        ix = reflect_clip(ix, (float)W);
        iy = reflect_clip(iy, (float)H);
        float x0f = floorf(ix), y0f = floorf(iy);
        float wx1 = ix - x0f,  wy1 = iy - y0f;
        int x0 = min(max((int)x0f, 0), W - 1);
        int y0 = min(max((int)y0f, 0), H - 1);
        int x1 = min(x0 + 1, W - 1);
        int y1 = min(y0 + 1, H - 1);
        int lx0 = x0 - (j0 - 3), lx1 = x1 - (j0 - 3);
        int ly0 = y0 - (i0 - 3), ly1 = y1 - (i0 - 3);
        float v00 = xp[(ly0 * XPITCH + lx0) * 2];
        float v01 = xp[(ly0 * XPITCH + lx1) * 2];
        float v10 = xp[(ly1 * XPITCH + lx0) * 2];
        float v11 = xp[(ly1 * XPITCH + lx1) * 2];
        float wx0 = 1.0f - wx1, wy0 = 1.0f - wy1;
        v = (v00 * wx0 + v01 * wx1) * wy0 + (v10 * wx0 + v11 * wx1) * wy1;
      }
      xdef[wave * NP + t] = v;
    }
    __syncthreads();

    const float* xd = xdef + wave * NP;
    #pragma unroll
    for (int oy = 0; oy < TY; ++oy) {
      float acc = 0.0f;
      #pragma unroll
      for (int dy = 0; dy < 3; ++dy)
        #pragma unroll
        for (int dx = 0; dx < 3; ++dx)
          acc += w9[dy * 3 + dx] * xd[(oy + dy) * EXW + (lane + dx)];
      acc = fmaxf(acc, 0.0f);
      out[((bb * CH + c) * H + (i0 + oy)) * W + (j0 + lane)] = acc;
    }
    __syncthreads();
  }
}

extern "C" void kernel_launch(void* const* d_in, const int* in_sizes, int n_in,
                              void* d_out, int out_size, void* d_ws, size_t ws_size,
                              hipStream_t stream) {
  const float* x        = (const float*)d_in[0];
  const float* offset_w = (const float*)d_in[1];
  const float* offset_b = (const float*)d_in[2];
  const float* dw_w     = (const float*)d_in[3];
  float* out = (float*)d_out;
  (void)in_sizes; (void)n_in; (void)out_size; (void)d_ws; (void)ws_size;
  dim3 grid(W / TX, H / TY, BATCH);   // 12 x 96 x 8
  dysnake_fused<<<grid, 256, 0, stream>>>(x, offset_w, offset_b, dw_w, out);
}
